// MixingBlock_87883620811281
// MI455X (gfx1250) — matrix operand order repaired and checked by reading, not compiled, not run on hardware
//
#include <hip/hip_runtime.h>
#include <hip/hip_bf16.h>

typedef __attribute__((ext_vector_type(16))) __bf16 v16bf;
typedef __attribute__((ext_vector_type(8)))  float  v8f;

union Frag16 {
  uint4 u[2];
  unsigned short h[16];
  v16bf v;
};

#define B_DIM  8
#define I_DIM  8192
#define S_DIM  64
#define IN_DIM 180
#define IN_PAD 192
#define D_DIM  1536
#define TILE_I 128

__device__ __forceinline__ unsigned short f32_to_bf16(float f) {
  unsigned int u = __float_as_uint(f);
  u += 0x7FFFu + ((u >> 16) & 1u);           // round-to-nearest-even
  return (unsigned short)(u >> 16);
}

__device__ __forceinline__ v8f wmma_bf16(const Frag16& a, const Frag16& b, v8f c) {
  return __builtin_amdgcn_wmma_f32_16x16x32_bf16(false, a.v, false, b.v,
                                                 (short)0, c, false, false);
}

// ---------------- precompute: f32 -> bf16 (with optional row padding) -------
__global__ void cvt_bf16_pad(const float* __restrict__ src,
                             unsigned short* __restrict__ dst,
                             int rows_src, int rows_dst, int cols) {
  int idx = blockIdx.x * blockDim.x + threadIdx.x;
  if (idx >= rows_dst * cols) return;
  int r = idx / cols;
  dst[idx] = (r < rows_src) ? f32_to_bf16(src[idx]) : (unsigned short)0;
}

// slotsT[b][d][s] = bf16(slots[b][s][d])
__global__ void slots_T_bf16(const float* __restrict__ slots,
                             unsigned short* __restrict__ outT) {
  int idx = blockIdx.x * blockDim.x + threadIdx.x;
  if (idx >= B_DIM * D_DIM * S_DIM) return;
  int b = idx / (D_DIM * S_DIM);
  int rem = idx % (D_DIM * S_DIM);
  int d = rem / S_DIM, s = rem % S_DIM;
  outT[idx] = f32_to_bf16(slots[((size_t)b * S_DIM + s) * D_DIM + d]);
}

// ---------------- precompute GEMM: C[m][n] = sum_k A[m][k]*B[n][k] ----------
// A: [M][K] bf16 row-major, B: [N][K] bf16 row-major, C: [M][N] bf16.
// One wave per 16x16 output tile. K multiple of 32.
__global__ __launch_bounds__(32) void gemm_nt_bf16(
    const unsigned short* __restrict__ A, const unsigned short* __restrict__ B,
    unsigned short* __restrict__ C, int M, int N, int K,
    long long strideA, long long strideB, long long strideC) {
  int tiles_n = N >> 4;
  int tm = (blockIdx.x / tiles_n) << 4;
  int tn = (blockIdx.x % tiles_n) << 4;
  const unsigned short* Ab = A + (long long)blockIdx.y * strideA;
  const unsigned short* Bb = B + (long long)blockIdx.y * strideB;
  unsigned short*       Cb = C + (long long)blockIdx.y * strideC;
  int lane = threadIdx.x;
  int r   = lane & 15;
  int khA = (lane >> 4) << 3;   // A: halves {khA..khA+7, 16+khA..}
  int khB = (lane >> 4) << 4;   // unused: B loads use khA, same halves as A
  v8f acc = {0.f,0.f,0.f,0.f,0.f,0.f,0.f,0.f};
  for (int k0 = 0; k0 < K; k0 += 32) {
    Frag16 fa, fb;
    const unsigned short* pa = Ab + (size_t)(tm + r) * K + k0 + khA;
    fa.u[0] = *(const uint4*)pa;
    fa.u[1] = *(const uint4*)(pa + 16);
    const unsigned short* pb = Bb + (size_t)(tn + r) * K + k0 + khA;
    fb.u[0] = *(const uint4*)pb;
    fb.u[1] = *(const uint4*)(pb + 16);
    acc = wmma_bf16(fa, fb, acc);
  }
  int cm = tm + ((lane >> 4) << 3);
  #pragma unroll
  for (int v = 0; v < 8; v++)
    Cb[(size_t)(cm + v) * N + (tn + r)] = f32_to_bf16(acc[v]);
}

// ---------------- fused main kernel -----------------------------------------
#define SMEM_SLOTS_OFF 0          // [1536][64] bf16 = 196608 B
#define SMEM_XN_OFF    196608     // [128][192] bf16 =  49152 B
#define SMEM_MT_OFF    245760     // [64][192]  bf16 =  24576 B
#define SMEM_WBUF_OFF  270336     // [8][16][64] f32 =  32768 B
#define SMEM_TOTAL     303104

__global__ __launch_bounds__(256, 1) void mixing_main(
    const float* __restrict__ x,
    const float* __restrict__ g1, const float* __restrict__ b1,
    const float* __restrict__ g2, const float* __restrict__ b2,
    const unsigned short* __restrict__ mtg,      // [B][64][192] bf16
    const unsigned short* __restrict__ slotsTg,  // [B][1536][64] bf16
    float* __restrict__ s_out, float* __restrict__ w_out) {
  extern __shared__ char smem[];
  unsigned short* sT = (unsigned short*)(smem + SMEM_SLOTS_OFF);
  unsigned short* xn = (unsigned short*)(smem + SMEM_XN_OFF);
  unsigned short* mt = (unsigned short*)(smem + SMEM_MT_OFF);
  float* wb_all      = (float*)(smem + SMEM_WBUF_OFF);

  const int b   = blockIdx.x >> 6;            // 64 tiles per batch
  const int i0  = (blockIdx.x & 63) * TILE_I;
  const int tid = threadIdx.x;

  // cooperative LDS fills
  {
    const uint4* g = (const uint4*)(slotsTg + (size_t)b * D_DIM * S_DIM);
    uint4* l = (uint4*)sT;
    for (int i = tid; i < (D_DIM * S_DIM) / 8; i += 256) l[i] = g[i];
  }
  {
    const uint4* g = (const uint4*)(mtg + (size_t)b * S_DIM * IN_PAD);
    uint4* l = (uint4*)mt;
    for (int i = tid; i < (S_DIM * IN_PAD) / 8; i += 256) l[i] = g[i];
  }

  // LayerNorm of x rows -> bf16 LDS (one thread per row)
  if (tid < TILE_I) {
    const float* xr = x + ((size_t)b * I_DIM + i0 + tid) * IN_DIM;
    float s = 0.f, sq = 0.f;
    for (int k = 0; k < IN_DIM; k++) { float v = xr[k]; s += v; sq += v * v; }
    float mu  = s * (1.f / IN_DIM);
    float var = sq * (1.f / IN_DIM) - mu * mu;
    float rs  = rsqrtf(var + 1e-5f);
    unsigned short* xo = xn + tid * IN_PAD;
    for (int k = 0; k < IN_DIM; k++)
      xo[k] = f32_to_bf16((xr[k] - mu) * rs * g1[k] + b1[k]);
    for (int k = IN_DIM; k < IN_PAD; k++) xo[k] = 0;
  }
  __syncthreads();

  const int wave = tid >> 5;
  const int lane = tid & 31;
  const int r0   = wave << 4;
  const int lr   = lane & 15;
  const int khA  = (lane >> 4) << 3;
  const int khB  = (lane >> 4) << 4;
  const int mh   = (lane >> 4) << 3;   // C-layout row offset

  // ---- dots (16x64) = xn_tile(16x192) @ Mt^T, K padded to 192
  v8f dacc[4];
  v8f zero = {0.f,0.f,0.f,0.f,0.f,0.f,0.f,0.f};
  #pragma unroll
  for (int t = 0; t < 4; t++) dacc[t] = zero;
  #pragma unroll
  for (int kk = 0; kk < 6; kk++) {
    Frag16 fa;
    const unsigned short* pa = xn + (r0 + lr) * IN_PAD + kk * 32 + khA;
    fa.u[0] = *(const uint4*)pa;
    fa.u[1] = *(const uint4*)(pa + 16);
    #pragma unroll
    for (int t = 0; t < 4; t++) {
      Frag16 fb;
      const unsigned short* pb = mt + (t * 16 + lr) * IN_PAD + kk * 32 + khA;
      fb.u[0] = *(const uint4*)pb;
      fb.u[1] = *(const uint4*)(pb + 16);
      dacc[t] = wmma_bf16(fa, fb, dacc[t]);
    }
  }

  // ---- softmax over S=64 (16-lane-half shfl reductions; wave32)
  const float scale = 0.025515518153991442f;   // 1536^-0.5
  float mx[8], sm[8];
  #pragma unroll
  for (int v = 0; v < 8; v++)
    mx[v] = scale * fmaxf(fmaxf(dacc[0][v], dacc[1][v]),
                          fmaxf(dacc[2][v], dacc[3][v]));
  #pragma unroll
  for (int v = 0; v < 8; v++)
    #pragma unroll
    for (int off = 1; off < 16; off <<= 1)
      mx[v] = fmaxf(mx[v], __shfl_xor(mx[v], off, 32));
  #pragma unroll
  for (int v = 0; v < 8; v++) sm[v] = 0.f;
  #pragma unroll
  for (int t = 0; t < 4; t++)
    #pragma unroll
    for (int v = 0; v < 8; v++) {
      float e = __expf(dacc[t][v] * scale - mx[v]);
      dacc[t][v] = e;
      sm[v] += e;
    }
  #pragma unroll
  for (int v = 0; v < 8; v++)
    #pragma unroll
    for (int off = 1; off < 16; off <<= 1)
      sm[v] += __shfl_xor(sm[v], off, 32);

  // stage w (f32) in LDS: transposes C-layout -> row-major
  float* wb = wb_all + wave * (16 * 64);
  #pragma unroll
  for (int v = 0; v < 8; v++) {
    float inv = 1.f / sm[v];
    #pragma unroll
    for (int t = 0; t < 4; t++)
      wb[(mh + v) * 64 + t * 16 + lr] = dacc[t][v] * inv;
  }

  // coalesced w output
  {
    float* wg = w_out + ((size_t)b * I_DIM + i0 + r0) * S_DIM;
    for (int idx = lane; idx < 16 * 64; idx += 32) wg[idx] = wb[idx];
  }

  // A-fragments of w (K=64 -> 2 frags), reused for both passes
  Frag16 wa[2];
  #pragma unroll
  for (int kk = 0; kk < 2; kk++)
    #pragma unroll
    for (int j = 0; j < 8; j++) {
      wa[kk].h[j]     = f32_to_bf16(wb[lr * 64 + kk * 32 + khA + j]);
      wa[kk].h[j + 8] = f32_to_bf16(wb[lr * 64 + kk * 32 + khA + 16 + j]);
    }

  // ---- s = w @ slots, pass 1: row stats only (recompute beats spilling HBM)
  float psum[8], psq[8];
  #pragma unroll
  for (int v = 0; v < 8; v++) { psum[v] = 0.f; psq[v] = 0.f; }
  #pragma unroll 2
  for (int nt = 0; nt < 96; nt++) {
    v8f acc = zero;
    #pragma unroll
    for (int kk = 0; kk < 2; kk++) {
      Frag16 fb;
      const unsigned short* pb = sT + (nt * 16 + lr) * S_DIM + kk * 32 + khA;
      fb.u[0] = *(const uint4*)pb;
      fb.u[1] = *(const uint4*)(pb + 16);
      acc = wmma_bf16(wa[kk], fb, acc);
    }
    #pragma unroll
    for (int v = 0; v < 8; v++) { psum[v] += acc[v]; psq[v] += acc[v] * acc[v]; }
  }
  #pragma unroll
  for (int v = 0; v < 8; v++)
    #pragma unroll
    for (int off = 1; off < 16; off <<= 1) {
      psum[v] += __shfl_xor(psum[v], off, 32);
      psq[v]  += __shfl_xor(psq[v],  off, 32);
    }
  float mean[8], rsg[8];
  #pragma unroll
  for (int v = 0; v < 8; v++) {
    float mu  = psum[v] * (1.f / D_DIM);
    float var = psq[v] * (1.f / D_DIM) - mu * mu;
    mean[v] = mu;
    rsg[v]  = rsqrtf(var + 1e-5f);
  }

  // ---- pass 2: recompute, layernorm, store
  float* sg = s_out + ((size_t)b * I_DIM + i0 + r0) * D_DIM;
  #pragma unroll 2
  for (int nt = 0; nt < 96; nt++) {
    v8f acc = zero;
    #pragma unroll
    for (int kk = 0; kk < 2; kk++) {
      Frag16 fb;
      const unsigned short* pb = sT + (nt * 16 + lr) * S_DIM + kk * 32 + khA;
      fb.u[0] = *(const uint4*)pb;
      fb.u[1] = *(const uint4*)(pb + 16);
      acc = wmma_bf16(wa[kk], fb, acc);
    }
    int n = nt * 16 + lr;
    float gv = g2[n], bv = b2[n];
    #pragma unroll
    for (int v = 0; v < 8; v++)
      sg[(size_t)(mh + v) * D_DIM + n] = (acc[v] - mean[v]) * rsg[v] * gv + bv;
  }
}

// ---------------- host launch ------------------------------------------------
extern "C" void kernel_launch(void* const* d_in, const int* in_sizes, int n_in,
                              void* d_out, int out_size, void* d_ws, size_t ws_size,
                              hipStream_t stream) {
  (void)in_sizes; (void)n_in; (void)out_size; (void)ws_size;
  const float* x     = (const float*)d_in[0];
  const float* slots = (const float*)d_in[1];
  const float* Wq    = (const float*)d_in[2];
  const float* Wk    = (const float*)d_in[3];
  const float* g1    = (const float*)d_in[4];
  const float* b1    = (const float*)d_in[5];
  const float* g2    = (const float*)d_in[6];
  const float* b2    = (const float*)d_in[7];
  float* s_out = (float*)d_out;
  float* w_out = s_out + (size_t)B_DIM * I_DIM * D_DIM;

  char* ws = (char*)d_ws;
  unsigned short* Wq_bf     = (unsigned short*)(ws + 0);        // 192*1536
  unsigned short* Wk_bf     = (unsigned short*)(ws + 589824);   // 1536*1536
  unsigned short* slots_bf  = (unsigned short*)(ws + 5308416);  // 8*64*1536
  unsigned short* slotsT_bf = (unsigned short*)(ws + 6881280);  // 8*1536*64
  unsigned short* Gt        = (unsigned short*)(ws + 8454144);  // 192*1536
  unsigned short* Mt        = (unsigned short*)(ws + 9043968);  // 8*64*192

  cvt_bf16_pad<<<(192 * 1536 + 255) / 256, 256, 0, stream>>>(Wq, Wq_bf, 180, 192, 1536);
  cvt_bf16_pad<<<(1536 * 1536 + 255) / 256, 256, 0, stream>>>(Wk, Wk_bf, 1536, 1536, 1536);
  cvt_bf16_pad<<<(512 * 1536 + 255) / 256, 256, 0, stream>>>(slots, slots_bf, 512, 512, 1536);
  slots_T_bf16<<<(B_DIM * D_DIM * S_DIM + 255) / 256, 256, 0, stream>>>(slots, slotsT_bf);

  // Gt[p][e] = sum_d Wq[p][d] * Wk[e][d]         (192 x 1536, K = 1536)
  gemm_nt_bf16<<<dim3((192 / 16) * (1536 / 16), 1), 32, 0, stream>>>(
      Wq_bf, Wk_bf, Gt, 192, 1536, 1536, 0, 0, 0);
  // Mt[b][s][p] = sum_e slots[b][s][e] * Gt[p][e] (64 x 192, K = 1536, 8 batches)
  gemm_nt_bf16<<<dim3((64 / 16) * (192 / 16), B_DIM), 32, 0, stream>>>(
      slots_bf, Gt, Mt, 64, 192, 1536,
      (long long)64 * 1536, 0, (long long)64 * 192);

  hipFuncSetAttribute((const void*)mixing_main,
                      hipFuncAttributeMaxDynamicSharedMemorySize, SMEM_TOTAL);
  mixing_main<<<dim3(B_DIM * (I_DIM / TILE_I)), 256, SMEM_TOTAL, stream>>>(
      x, g1, b1, g2, b2, Mt, slotsT_bf, s_out, w_out);
}
